// AdvancedChimeraLayer_9234179686568
// MI455X (gfx1250) — matrix operand order repaired and checked by reading, not compiled, not run on hardware
//
#include <hip/hip_runtime.h>
#include <hip/hip_bf16.h>
#include <math.h>

// ---------------- problem constants (from reference) ----------------
#define B_    4
#define S_    2048
#define DM    1024      // D_MODEL
#define DI    2048      // D_INNER
#define HN    32        // NHEADS
#define HP    64        // HEADDIM
#define DS    128       // D_STATE
#define CDIM  2304      // CONV_DIM
#define DPROJ 4384      // D_IN_PROJ
#define NPAD_IP 4480    // DPROJ padded to multiple of 128
#define NC    8         // NCACHE
#define BUS   128

typedef __attribute__((ext_vector_type(16))) _Float16 v16h;
typedef __attribute__((ext_vector_type(8)))  _Float16 v8h;
typedef __attribute__((ext_vector_type(8)))  float    v8f;
typedef __attribute__((ext_vector_type(4))) unsigned int u32x4;
typedef __attribute__((ext_vector_type(8))) int          i32x8;
typedef __attribute__((ext_vector_type(4))) int          i32x4;

__device__ __forceinline__ float silu_f(float x) { return x / (1.f + expf(-x)); }

// ---------------- Tensor Data Mover: 2D f32 tile (tile_w x tile_h) global -> LDS ----------------
// D# built per CDNA5 ISA ch.8: group0 = {count/type/addr}, group1 = {data_size, dims, tile, stride}.
// Tensor dims set huge (2^30) so no OOB clipping occurs; global_addr is the tile start.
// This toolchain exposes the 6-arg builtin: (g0, g1, g2, g3, g4, cpol).
__device__ __forceinline__ void tdm_load_2d_f32(unsigned lds_addr, const float* gptr,
                                                unsigned tile_w, unsigned tile_h,
                                                unsigned row_stride_elts) {
  unsigned long long ga = (unsigned long long)(size_t)gptr;
  u32x4 g0;
  g0[0] = 1u;                                                  // count=1 (valid), user mode
  g0[1] = lds_addr;                                            // lds_addr (bytes)
  g0[2] = (unsigned)ga;                                        // global_addr[31:0]
  g0[3] = (unsigned)((ga >> 32) & 0x01ffffffu) | (2u << 30);   // global_addr[56:32] | type=2
  i32x8 g1;
  g1[0] = (2 << 16);                         // workgroup_mask=0, data_size=2 (4-byte)
  g1[1] = 0;                                 // atomic_barrier=0, tensor_dim0[15:0]=0
  g1[2] = 0x4000;                            // tensor_dim0[31:16]=0x4000 (2^30), tensor_dim1 lo=0
  g1[3] = 0x4000 | (int)(tile_w << 16);      // tensor_dim1[31:16]=0x4000 (2^30), tile_dim0
  g1[4] = (int)tile_h;                       // tile_dim1; tile_dim2=0
  g1[5] = (int)row_stride_elts;              // tensor_dim0_stride[31:0] (data_size units)
  g1[6] = 0;                                 // stride hi, dim1_stride lo
  g1[7] = 0;
  i32x4 z4 = {0, 0, 0, 0};
  i32x8 z8 = {0, 0, 0, 0, 0, 0, 0, 0};
  __builtin_amdgcn_tensor_load_to_lds(g0, g1, z4, z4, z8, 0);
}

// ---------------- RMSNorm of x -> f16 activations + per-row scale ----------------
__global__ __launch_bounds__(256)
void rmsnorm_kernel(const float* __restrict__ x, const float* __restrict__ norm_w,
                    _Float16* __restrict__ xnf16, float* __restrict__ rowscale) {
  __shared__ float red[256];
  int row = blockIdx.x;                 // B*S rows
  int t = threadIdx.x;
  const float* xr = x + (size_t)row * DM;
  float v[4]; float acc = 0.f;
#pragma unroll
  for (int i = 0; i < 4; i++) { v[i] = xr[t + 256 * i]; acc += v[i] * v[i]; }
  red[t] = acc; __syncthreads();
  for (int off = 128; off > 0; off >>= 1) { if (t < off) red[t] += red[t + off]; __syncthreads(); }
  float scale = rsqrtf(red[0] / (float)DM + 1e-5f);
  if (t == 0) rowscale[row] = scale;
#pragma unroll
  for (int i = 0; i < 4; i++) {
    int d = t + 256 * i;
    xnf16[(size_t)row * DM + d] = (_Float16)(v[i] * scale * norm_w[d]);
  }
}

// ---------------- mean over seq of x_norm (reconstructed via rowscale) ----------------
__global__ void mean_xnorm_kernel(const float* __restrict__ x, const float* __restrict__ rowscale,
                                  const float* __restrict__ norm_w, float* __restrict__ meanx) {
  int d = blockIdx.x * blockDim.x + threadIdx.x;  // 0..DM-1
  int b = blockIdx.y;
  float acc = 0.f;
  for (int s = 0; s < S_; s++) acc += x[((size_t)(b * S_ + s)) * DM + d] * rowscale[b * S_ + s];
  meanx[b * DM + d] = acc * (1.f / (float)S_) * norm_w[d];
}

// ---------------- router MLP + tempered softmax -> probs output ----------------
__global__ void router_kernel(const float* __restrict__ meanx,
                              const float* __restrict__ r_w1, const float* __restrict__ r_b1,
                              const float* __restrict__ r_w2, const float* __restrict__ r_b2,
                              const float* __restrict__ log_temp, float* __restrict__ out_probs) {
  __shared__ float hbuf[32];
  __shared__ float lg[3];
  int t = threadIdx.x;  // 32 threads
  for (int b = 0; b < B_; b++) {
    float acc = r_b1[t];
    const float* w = r_w1 + t * DM;
    const float* m = meanx + b * DM;
    for (int d = 0; d < DM; d++) acc += w[d] * m[d];
    hbuf[t] = silu_f(acc);
    __syncthreads();
    if (t < 3) {
      float l = r_b2[t];
      for (int j = 0; j < 32; j++) l += r_w2[t * 32 + j] * hbuf[j];
      lg[t] = l;
    }
    __syncthreads();
    if (t == 0) {
      float T = fminf(fmaxf(expf(log_temp[0]), 0.1f), 10.f);
      float a0 = lg[0] / T, a1 = lg[1] / T, a2 = lg[2] / T;
      float mx = fmaxf(a0, fmaxf(a1, a2));
      float e0 = expf(a0 - mx), e1 = expf(a1 - mx), e2 = expf(a2 - mx);
      float den = e0 + e1 + e2;
      float p0 = fmaxf(e0 / den, 0.05f), p1 = fmaxf(e1 / den, 0.05f), p2 = fmaxf(e2 / den, 0.05f);
      float dn2 = p0 + p1 + p2;
      out_probs[b * 3 + 0] = p0 / dn2;
      out_probs[b * 3 + 1] = p1 / dn2;
      out_probs[b * 3 + 2] = p2 / dn2;
    }
    __syncthreads();
  }
}

// ---------------- fp32 -> f16 weight conversion with optional row padding ----------------
__global__ void cvt_f16_pad_kernel(const float* __restrict__ src, _Float16* __restrict__ dst,
                                   long rows, long cols, long padRows) {
  long idx = (long)blockIdx.x * blockDim.x + threadIdx.x;
  if (idx >= padRows * cols) return;
  long r = idx / cols;
  dst[idx] = (r < rows) ? (_Float16)src[idx] : (_Float16)0.f;
}

// ---------------- generic WMMA GEMM: C = addend + alpha * A(MxK,f16) @ Bw(NxK,f16)^T ----------------
// Block tile 64x128, 8 waves in 2(M)x4(N) grid, each wave 2x2 fragments of 16x16, BK=32.
// Double-buffered LDS: one barrier per K-step, tile k+1 staged under the WMMAs of tile k.
__global__ __launch_bounds__(256)
void gemm_wmma_f16_kernel(const _Float16* __restrict__ A, const _Float16* __restrict__ Bw,
                          const float* __restrict__ addend, const float* __restrict__ alpha_ptr,
                          float* __restrict__ C, int M, int N, int K) {
  __shared__ __align__(16) _Float16 sA[2][64][32];    // [buf][m][k]
  __shared__ __align__(16) _Float16 sB[2][128][32];   // [buf][n][k]
  int t = threadIdx.x;
  int lane = t & 31;
  int wave = t >> 5;
  int wm = wave & 1;        // 2 waves along M (32 rows each)
  int wn = wave >> 1;       // 4 waves along N (32 cols each)
  int m0 = blockIdx.x * 64;
  int n0 = blockIdx.y * 128;
  const v8f vzero = {0.f, 0.f, 0.f, 0.f, 0.f, 0.f, 0.f, 0.f};
  v8f acc[2][2];
  acc[0][0] = vzero; acc[0][1] = vzero; acc[1][0] = vzero; acc[1][1] = vzero;

  int arow = t >> 2, acol = (t & 3) * 8;   // A tile: 64 rows x 32 halves, 16B/thread
  int brow = t >> 1, bcol = (t & 1) * 16;  // B tile: 128 rows x 32 halves, 32B/thread
  const int ml = lane & 15, hi = lane >> 4;
  const _Float16* gA = A + (size_t)(m0 + arow) * K + acol;
  const _Float16* gB = Bw + (size_t)(n0 + brow) * K + bcol;

  auto stage = [&](int buf, int k0) {
    *(v8h*)&sA[buf][arow][acol]   = *(const v8h*)(gA + k0);
    *(v8h*)&sB[buf][brow][bcol]   = *(const v8h*)(gB + k0);
    *(v8h*)&sB[buf][brow][bcol+8] = *(const v8h*)(gB + k0 + 8);
  };

  stage(0, 0);
  __syncthreads();

  for (int k0 = 0; k0 < K; k0 += 32) {
    int buf = (k0 >> 5) & 1;
    if (k0 + 32 < K) {
      stage(buf ^ 1, k0 + 32);
      if (k0 + 64 < K) {   // near-cache prefetch of the tile after next
        __builtin_prefetch(gA + k0 + 64, 0, 3);
        __builtin_prefetch(gB + k0 + 64, 0, 3);
      }
    }
    v16h af[2], bf[2];
#pragma unroll
    for (int i = 0; i < 2; i++) {
      int r = wm * 32 + i * 16 + ml;
      v8h lo = *(const v8h*)&sA[buf][r][hi * 8];        // K = hi*8 .. +7
      v8h hh = *(const v8h*)&sA[buf][r][16 + hi * 8];   // K = 16+hi*8 .. +7
      af[i] = __builtin_shufflevector(lo, hh, 0, 1, 2, 3, 4, 5, 6, 7, 8, 9, 10, 11, 12, 13, 14, 15);
    }
#pragma unroll
    for (int j = 0; j < 2; j++) {
      int c = wn * 32 + j * 16 + ml;
      v8h lo = *(const v8h*)&sB[buf][c][hi * 8];        // K = hi*8 .. +7
      v8h hh = *(const v8h*)&sB[buf][c][16 + hi * 8];   // K = 16+hi*8 .. +7
      bf[j] = __builtin_shufflevector(lo, hh, 0, 1, 2, 3, 4, 5, 6, 7, 8, 9, 10, 11, 12, 13, 14, 15);
    }
#pragma unroll
    for (int i = 0; i < 2; i++)
#pragma unroll
      for (int j = 0; j < 2; j++)
        acc[i][j] = __builtin_amdgcn_wmma_f32_16x16x32_f16(
            false, af[i], false, bf[j], (short)0, acc[i][j], false, false);
    __syncthreads();
  }

  float alpha = alpha_ptr ? alpha_ptr[0] : 1.0f;
#pragma unroll
  for (int i = 0; i < 2; i++)
#pragma unroll
    for (int j = 0; j < 2; j++)
#pragma unroll
      for (int r = 0; r < 8; r++) {
        int m = m0 + wm * 32 + i * 16 + hi * 8 + r;   // C/D: VGPR r -> M=r(+8 for hi lanes)
        int n = n0 + wn * 32 + j * 16 + ml;
        if (n < N) {
          float v = acc[i][j][r] * alpha;
          if (addend) v += addend[(size_t)m * N + n];
          C[(size_t)m * N + n] = v;
        }
      }
}

// ---------------- depthwise causal conv(4) + bias + SiLU ----------------
__global__ void conv_silu_kernel(const float* __restrict__ zx, const float* __restrict__ conv_w,
                                 const float* __restrict__ conv_b, float* __restrict__ xbcc) {
  long idx = (long)blockIdx.x * blockDim.x + threadIdx.x;
  if (idx >= (long)B_ * S_ * CDIM) return;
  int c = (int)(idx % CDIM);
  long bs = idx / CDIM;
  int s = (int)(bs % S_);
  long b = bs / S_;
  float acc = conv_b[c];
#pragma unroll
  for (int j = 0; j < 4; j++) {
    int sp = s - 3 + j;
    if (sp >= 0) acc += conv_w[c * 4 + j] * zx[((b * S_ + sp) * (long)DPROJ) + DI + c];
  }
  xbcc[idx] = silu_f(acc);
}

// ---------------- dt softplus + decay dA = exp(dt * A) ----------------
__global__ void dt_kernel(const float* __restrict__ zx, const float* __restrict__ dt_bias,
                          const float* __restrict__ A_log, float* __restrict__ dtp,
                          float* __restrict__ dab) {
  long idx = (long)blockIdx.x * blockDim.x + threadIdx.x;
  if (idx >= (long)B_ * S_ * HN) return;
  int h = (int)(idx % HN);
  long bs = idx / HN;
  float raw = zx[bs * (long)DPROJ + (DPROJ - HN) + h] + dt_bias[h];
  float dtv = (raw > 20.f) ? raw : log1pf(expf(raw));
  dtp[idx] = dtv;
  dab[idx] = expf(dtv * (-expf(A_log[h])));
}

// ---------------- selective-state scan: one WGP per (batch, head); state in VGPRs ----------------
// B/C/x rows for 8 timesteps are fetched by the Tensor Data Mover (async, double-buffered):
// the DMA for chunk c+1 runs underneath the 8 serial scan steps of chunk c.
#define SCHUNK 8
__global__ __launch_bounds__(256)
void ssm_scan_kernel(const float* __restrict__ xbcc, const float* __restrict__ dtp,
                     const float* __restrict__ dab, const float* __restrict__ D_param,
                     float* __restrict__ y) {
  __shared__ __align__(16) float sBC[2][SCHUNK][2 * DS];  // per step: [B(128) | C(128)] contiguous
  __shared__ __align__(16) float sx[2][SCHUNK][HP];
  __shared__ float sa[2][SCHUNK];
  __shared__ float sd[2][SCHUNK];
  int bh = blockIdx.x;
  int b = bh >> 5, h = bh & 31;
  int t = threadIdx.x;
  int p = t >> 2, qq = t & 3, nb = qq * 32;  // thread owns (p, n = nb..nb+31)
  float st[32];
#pragma unroll
  for (int k = 0; k < 32; k++) st[k] = 0.f;
  float Dp = D_param[h];
  const size_t base = (size_t)b * S_;

  auto issue = [&](int buf, int s0) {
    if (t < 32) {  // wave 0 issues the TDM descriptors (EXEC-independent, per-wave TENSORcnt)
      const float* gBC = xbcc + (base + s0) * (size_t)CDIM + DI;       // B|C: 256 floats / step
      tdm_load_2d_f32((unsigned)(size_t)&sBC[buf][0][0], gBC, 2 * DS, SCHUNK, CDIM);
      const float* gx = xbcc + (base + s0) * (size_t)CDIM + h * HP;    // x head slice: 64 / step
      tdm_load_2d_f32((unsigned)(size_t)&sx[buf][0][0], gx, HP, SCHUNK, CDIM);
    }
    if (t < SCHUNK)          sa[buf][t] = dab[(base + s0 + t) * HN + h];
    else if (t < 2 * SCHUNK) sd[buf][t - SCHUNK] = dtp[(base + s0 + (t - SCHUNK)) * HN + h];
  };

  issue(0, 0);
  if (t < 32) __builtin_amdgcn_s_wait_tensorcnt(0);
  __syncthreads();

  for (int s0 = 0; s0 < S_; s0 += SCHUNK) {
    int buf = (s0 / SCHUNK) & 1;
    if (s0 + SCHUNK < S_) issue(buf ^ 1, s0 + SCHUNK);   // overlap DMA with compute
#pragma unroll 1
    for (int ss = 0; ss < SCHUNK; ss++) {
      float a = sa[buf][ss];
      float coef = sd[buf][ss] * sx[buf][ss][p];
      float yacc = 0.f;
#pragma unroll
      for (int k = 0; k < 32; k++) {
        st[k] = a * st[k] + coef * sBC[buf][ss][nb + k];
        yacc += st[k] * sBC[buf][ss][DS + nb + k];
      }
      yacc += __shfl_xor(yacc, 1);
      yacc += __shfl_xor(yacc, 2);
      if (qq == 0)
        y[(base + s0 + ss) * DI + h * HP + p] = yacc + Dp * sx[buf][ss][p];
    }
    __syncthreads();                       // all waves done reading buf
    if (s0 + SCHUNK < S_) {
      if (t < 32) __builtin_amdgcn_s_wait_tensorcnt(0);  // buf^1 DMA complete
      __syncthreads();                     // publish to all waves
    }
  }
}

// ---------------- gated RMSNorm: ygn = rms(y * silu(z)) * gnorm_w, output f16 ----------------
__global__ __launch_bounds__(256)
void gated_rmsnorm_kernel(const float* __restrict__ y, const float* __restrict__ zx,
                          const float* __restrict__ gnorm_w, _Float16* __restrict__ ygn) {
  __shared__ float red[256];
  long row = blockIdx.x;
  int t = threadIdx.x;
  const float* yr = y + row * (long)DI;
  const float* zr = zx + row * (long)DPROJ;   // z = first DI channels of zxbcdt
  float vals[8]; float acc = 0.f;
#pragma unroll
  for (int i = 0; i < 8; i++) {
    int e = t + 256 * i;
    float v = yr[e] * silu_f(zr[e]);
    vals[i] = v; acc += v * v;
  }
  red[t] = acc; __syncthreads();
  for (int off = 128; off > 0; off >>= 1) { if (t < off) red[t] += red[t + off]; __syncthreads(); }
  float scale = rsqrtf(red[0] / (float)DI + 1e-5f);
#pragma unroll
  for (int i = 0; i < 8; i++) {
    int e = t + 256 * i;
    ygn[row * (long)DI + e] = (_Float16)(vals[i] * scale * gnorm_w[e]);
  }
}

// ---------------- mean over seq (generic, width DM) ----------------
__global__ void mean_seq_kernel(const float* __restrict__ src, float* __restrict__ dst) {
  int d = blockIdx.x * blockDim.x + threadIdx.x;
  int b = blockIdx.y;
  float acc = 0.f;
  for (int s = 0; s < S_; s++) acc += src[((size_t)(b * S_ + s)) * DM + d];
  dst[b * DM + d] = acc * (1.f / (float)S_);
}

// ---------------- summary = normalize(mean(x_res) @ pub_W^T) ----------------
__global__ void summary_kernel(const float* __restrict__ meanxr, const float* __restrict__ pub_W,
                               float* __restrict__ summary) {
  __shared__ float red[128];
  int t = threadIdx.x;  // 128 threads
  for (int b = 0; b < B_; b++) {
    float acc = 0.f;
    const float* w = pub_W + t * DM;
    const float* m = meanxr + b * DM;
    for (int d = 0; d < DM; d++) acc += w[d] * m[d];
    red[t] = acc * acc; __syncthreads();
    for (int off = 64; off > 0; off >>= 1) { if (t < off) red[t] += red[t + off]; __syncthreads(); }
    float nrm = fmaxf(sqrtf(red[0]), 1e-12f);
    summary[b * BUS + t] = acc / nrm;
    __syncthreads();
  }
}

// ---------------- new_cache output = concat(bus_cache, summary) ----------------
__global__ void cache_out_kernel(const float* __restrict__ bus_cache, const float* __restrict__ summary,
                                 float* __restrict__ out_cache) {
  int idx = blockIdx.x * blockDim.x + threadIdx.x;
  if (idx >= B_ * (NC + 1) * BUS) return;
  int d = idx % BUS;
  int n = (idx / BUS) % (NC + 1);
  int b = idx / (BUS * (NC + 1));
  out_cache[idx] = (n < NC) ? bus_cache[(b * NC + n) * BUS + d] : summary[b * BUS + d];
}

// ---------------- bus attention: softmax over 9 keys, gather -> f16 ----------------
__global__ __launch_bounds__(128)
void bus_attn_kernel(const float* __restrict__ q, const float* __restrict__ summary,
                     const float* __restrict__ bus_cache, _Float16* __restrict__ gath) {
  __shared__ float saug[9][BUS];
  __shared__ float red[BUS];
  __shared__ float slog[9];
  long bs = blockIdx.x;
  int b = (int)(bs / S_);
  int t = threadIdx.x;
  saug[0][t] = summary[b * BUS + t];
#pragma unroll
  for (int n = 1; n < 9; n++) saug[n][t] = bus_cache[((long)b * NC + (n - 1)) * BUS + t];
  __syncthreads();
  float qv = q[bs * BUS + t];
  for (int n = 0; n < 9; n++) {
    red[t] = qv * saug[n][t]; __syncthreads();
    for (int off = 64; off > 0; off >>= 1) { if (t < off) red[t] += red[t + off]; __syncthreads(); }
    if (t == 0) slog[n] = red[0] * 0.08838834764831845f;  // 1/sqrt(128)
    __syncthreads();
  }
  float mx = -1e30f;
#pragma unroll
  for (int n = 0; n < 9; n++) mx = fmaxf(mx, slog[n]);
  float den = 0.f, g = 0.f;
#pragma unroll
  for (int n = 0; n < 9; n++) { float w = expf(slog[n] - mx); den += w; g += w * saug[n][t]; }
  gath[bs * BUS + t] = (_Float16)(g / den);
}

__global__ void sigmoid_gate_kernel(const float* __restrict__ gate, float* __restrict__ alpha) {
  if (threadIdx.x == 0 && blockIdx.x == 0) alpha[0] = 1.f / (1.f + expf(-gate[0]));
}

// ---------------- workspace layout ----------------
static constexpr size_t ALGN = 256;
static constexpr size_t al_(size_t x) { return (x + ALGN - 1) & ~(ALGN - 1); }
static constexpr size_t OFF_XNF16   = 0;
static constexpr size_t OFF_ROWSC   = OFF_XNF16   + al_((size_t)B_ * S_ * DM * 2);
static constexpr size_t OFF_MEANX   = OFF_ROWSC   + al_((size_t)B_ * S_ * 4);
static constexpr size_t OFF_WIP     = OFF_MEANX   + al_((size_t)B_ * DM * 4);
static constexpr size_t OFF_WOP     = OFF_WIP     + al_((size_t)NPAD_IP * DM * 2);
static constexpr size_t OFF_WGQ     = OFF_WOP     + al_((size_t)DM * DI * 2);
static constexpr size_t OFF_WMOD    = OFF_WGQ     + al_((size_t)BUS * DM * 2);
static constexpr size_t OFF_ZX      = OFF_WMOD    + al_((size_t)DM * BUS * 2);
static constexpr size_t OFF_XBCC    = OFF_ZX      + al_((size_t)B_ * S_ * DPROJ * 4);
static constexpr size_t OFF_DTP     = OFF_XBCC    + al_((size_t)B_ * S_ * CDIM * 4);
static constexpr size_t OFF_DA      = OFF_DTP     + al_((size_t)B_ * S_ * HN * 4);
static constexpr size_t OFF_Y       = OFF_DA      + al_((size_t)B_ * S_ * HN * 4);
static constexpr size_t OFF_YGN     = OFF_Y       + al_((size_t)B_ * S_ * DI * 4);
static constexpr size_t OFF_XRES    = OFF_YGN     + al_((size_t)B_ * S_ * DI * 2);
static constexpr size_t OFF_XRESH   = OFF_XRES    + al_((size_t)B_ * S_ * DM * 4);
static constexpr size_t OFF_MEANXR  = OFF_XRESH   + al_((size_t)B_ * S_ * DM * 2);
static constexpr size_t OFF_SUM     = OFF_MEANXR  + al_((size_t)B_ * DM * 4);
static constexpr size_t OFF_Q       = OFF_SUM     + al_((size_t)B_ * BUS * 4);
static constexpr size_t OFF_GATH    = OFF_Q       + al_((size_t)B_ * S_ * BUS * 4);
static constexpr size_t OFF_ALPHA   = OFF_GATH    + al_((size_t)B_ * S_ * BUS * 2);

extern "C" void kernel_launch(void* const* d_in, const int* in_sizes, int n_in,
                              void* d_out, int out_size, void* d_ws, size_t ws_size,
                              hipStream_t stream) {
  const float* x          = (const float*)d_in[0];
  const float* bus_cache  = (const float*)d_in[1];
  const float* norm_w     = (const float*)d_in[2];
  const float* in_proj_W  = (const float*)d_in[3];
  const float* conv_w     = (const float*)d_in[4];
  const float* conv_b     = (const float*)d_in[5];
  const float* dt_bias    = (const float*)d_in[6];
  const float* A_log      = (const float*)d_in[7];
  const float* D_param    = (const float*)d_in[8];
  const float* gnorm_w    = (const float*)d_in[9];
  const float* out_proj_W = (const float*)d_in[10];
  const float* r_w1       = (const float*)d_in[11];
  const float* r_b1       = (const float*)d_in[12];
  const float* r_w2       = (const float*)d_in[13];
  const float* r_b2       = (const float*)d_in[14];
  const float* log_temp   = (const float*)d_in[15];
  const float* pub_W      = (const float*)d_in[16];
  const float* gq_W       = (const float*)d_in[17];
  const float* mod_W      = (const float*)d_in[18];
  const float* gate       = (const float*)d_in[19];

  float* out       = (float*)d_out;                            // (B,S,DM)
  float* out_cache = out + (size_t)B_ * S_ * DM;               // (B,9,BUS)
  float* out_probs = out_cache + (size_t)B_ * (NC + 1) * BUS;  // (B,3)

  char* ws = (char*)d_ws;
  _Float16* xnf16  = (_Float16*)(ws + OFF_XNF16);
  float*    rowsc  = (float*)   (ws + OFF_ROWSC);
  float*    meanx  = (float*)   (ws + OFF_MEANX);
  _Float16* Wip    = (_Float16*)(ws + OFF_WIP);
  _Float16* Wop    = (_Float16*)(ws + OFF_WOP);
  _Float16* Wgq    = (_Float16*)(ws + OFF_WGQ);
  _Float16* Wmod   = (_Float16*)(ws + OFF_WMOD);
  float*    zx     = (float*)   (ws + OFF_ZX);
  float*    xbcc   = (float*)   (ws + OFF_XBCC);
  float*    dtp    = (float*)   (ws + OFF_DTP);
  float*    dab    = (float*)   (ws + OFF_DA);
  float*    ybuf   = (float*)   (ws + OFF_Y);
  _Float16* ygn    = (_Float16*)(ws + OFF_YGN);
  float*    xres   = (float*)   (ws + OFF_XRES);
  _Float16* xresh  = (_Float16*)(ws + OFF_XRESH);
  float*    meanxr = (float*)   (ws + OFF_MEANXR);
  float*    summ   = (float*)   (ws + OFF_SUM);
  float*    qbuf   = (float*)   (ws + OFF_Q);
  _Float16* gath   = (_Float16*)(ws + OFF_GATH);
  float*    alpha  = (float*)   (ws + OFF_ALPHA);

  const long ROWS = (long)B_ * S_;   // 8192

  // 1) RMSNorm -> f16 activations + row scales
  rmsnorm_kernel<<<(int)ROWS, 256, 0, stream>>>(x, norm_w, xnf16, rowsc);
  // 2) seq-mean of x_norm; 3) router probs
  mean_xnorm_kernel<<<dim3(DM / 256, B_), 256, 0, stream>>>(x, rowsc, norm_w, meanx);
  router_kernel<<<1, 32, 0, stream>>>(meanx, r_w1, r_b1, r_w2, r_b2, log_temp, out_probs);
  // 4) weight conversions (fp32 -> f16, in_proj padded to 4480 rows)
  {
    long n1 = (long)NPAD_IP * DM, n2 = (long)DM * DI, n3 = (long)BUS * DM, n4 = (long)DM * BUS;
    cvt_f16_pad_kernel<<<(int)((n1 + 255) / 256), 256, 0, stream>>>(in_proj_W, Wip, DPROJ, DM, NPAD_IP);
    cvt_f16_pad_kernel<<<(int)((n2 + 255) / 256), 256, 0, stream>>>(out_proj_W, Wop, DM, DI, DM);
    cvt_f16_pad_kernel<<<(int)((n3 + 255) / 256), 256, 0, stream>>>(gq_W, Wgq, BUS, DM, BUS);
    cvt_f16_pad_kernel<<<(int)((n4 + 255) / 256), 256, 0, stream>>>(mod_W, Wmod, DM, BUS, DM);
  }
  // 5) in_proj GEMM: zxbcdt = x_norm @ in_proj_W^T  (M=8192, N=4384, K=1024)
  gemm_wmma_f16_kernel<<<dim3(ROWS / 64, NPAD_IP / 128), 256, 0, stream>>>(
      xnf16, Wip, nullptr, nullptr, zx, (int)ROWS, DPROJ, DM);
  // 6) depthwise conv + SiLU; 7) dt/dA
  {
    long n = (long)B_ * S_ * CDIM;
    conv_silu_kernel<<<(int)((n + 255) / 256), 256, 0, stream>>>(zx, conv_w, conv_b, xbcc);
    long m = (long)B_ * S_ * HN;
    dt_kernel<<<(int)((m + 255) / 256), 256, 0, stream>>>(zx, dt_bias, A_log, dtp, dab);
  }
  // 8) SSM scan: 128 blocks = (batch, head), TDM double-buffered staging
  ssm_scan_kernel<<<B_ * HN, 256, 0, stream>>>(xbcc, dtp, dab, D_param, ybuf);
  // 9) gated RMSNorm -> f16
  gated_rmsnorm_kernel<<<(int)ROWS, 256, 0, stream>>>(ybuf, zx, gnorm_w, ygn);
  // 10) out_proj GEMM + residual: x_res = x + ygn @ out_proj_W^T  (N=1024, K=2048)
  gemm_wmma_f16_kernel<<<dim3(ROWS / 64, DM / 128), 256, 0, stream>>>(
      ygn, Wop, x, nullptr, xres, (int)ROWS, DM, DI);
  // 11) summary head
  mean_seq_kernel<<<dim3(DM / 256, B_), 256, 0, stream>>>(xres, meanxr);
  summary_kernel<<<1, 128, 0, stream>>>(meanxr, pub_W, summ);
  cache_out_kernel<<<(B_ * (NC + 1) * BUS + 255) / 256, 256, 0, stream>>>(bus_cache, summ, out_cache);
  // 12) q = x_res @ gq_W^T  (N=128, K=1024)
  cvt_f16_pad_kernel<<<(int)((ROWS * DM + 255) / 256), 256, 0, stream>>>(xres, xresh, ROWS, DM, ROWS);
  gemm_wmma_f16_kernel<<<dim3(ROWS / 64, 1), 256, 0, stream>>>(
      xresh, Wgq, nullptr, nullptr, qbuf, (int)ROWS, BUS, DM);
  // 13) bus attention -> gathered (f16)
  bus_attn_kernel<<<(int)ROWS, 128, 0, stream>>>(qbuf, summ, bus_cache, gath);
  // 14) final GEMM: out = x_res + sigmoid(gate) * gathered @ mod_W^T  (N=1024, K=128)
  sigmoid_gate_kernel<<<1, 32, 0, stream>>>(gate, alpha);
  gemm_wmma_f16_kernel<<<dim3(ROWS / 64, DM / 128), 256, 0, stream>>>(
      gath, Wmod, xres, alpha, out, (int)ROWS, DM, BUS);
}
